// StageNet_16913581212077
// MI455X (gfx1250) — matrix operand order repaired and checked by reading, not compiled, not run on hardware
//
#include <hip/hip_runtime.h>
#include <hip/hip_bf16.h>

// ---------------- problem constants ----------------
#define Bsz   256
#define Tst   400
#define Dd    280
#define Hh    384
#define Ll    12
#define Kk    10
#define CHUNK 32          // H / L
#define Gg    1560        // 4H + 2L
#define H6    64          // H / 6
#define KU    666         // D+1 + H+1  (true contraction length)
#define KP    672         // KU padded to multiple of 32 (21 * 32)
#define NP    1792        // G padded to 7 * 256 (no guards in GEMM)

typedef __attribute__((ext_vector_type(16))) __bf16 v16bf;
typedef __attribute__((ext_vector_type(8)))  float  v8f;

union FragU { v16bf v; unsigned int u[8]; };

__device__ __forceinline__ unsigned short f32_to_bf16(float f) {
    unsigned int u = __float_as_uint(f);
    unsigned int r = u + 0x7FFFu + ((u >> 16) & 1u);   // round-to-nearest-even
    return (unsigned short)(r >> 16);
}

__device__ __forceinline__ float sigmoidf_(float x) {
    return 1.0f / (1.0f + __expf(-x));
}

// ---------------------------------------------------------------------------
// Kernel 0: convert [Wk ; Wr] -> zero-padded bf16 [KP][NP], bias = bk + br
// ---------------------------------------------------------------------------
__global__ void prep_weights(const float* __restrict__ Wk, const float* __restrict__ bk,
                             const float* __restrict__ Wr, const float* __restrict__ br,
                             unsigned short* __restrict__ WB, float* __restrict__ bias)
{
    int idx = blockIdx.x * blockDim.x + threadIdx.x;
    if (idx < KP * NP) {
        int k = idx / NP;
        int n = idx % NP;
        float v = 0.0f;
        if (n < Gg) {
            if (k < Dd + 1)      v = Wk[(size_t)k * Gg + n];
            else if (k < KU)     v = Wr[(size_t)(k - (Dd + 1)) * Gg + n];
        }
        WB[idx] = f32_to_bf16(v);
    }
    if (idx < NP) bias[idx] = (idx < Gg) ? (bk[idx] + br[idx]) : 0.0f;
}

// ---------------------------------------------------------------------------
// Kernel 1: zero-init recurrent state h, c, and the packed activation matrix
// (zeros give h=0 for step 0 and permanent zero K-padding in cols 666..671)
// ---------------------------------------------------------------------------
__global__ void init_state(float* __restrict__ h, float* __restrict__ c,
                           unsigned int* __restrict__ Abf_u32)
{
    int i = blockIdx.x * blockDim.x + threadIdx.x;
    if (i < Bsz * Hh) { h[i] = 0.0f; c[i] = 0.0f; }
    if (i < (Bsz * KP) / 2) Abf_u32[i] = 0u;
}

// ---------------------------------------------------------------------------
// Kernel 2 (per step): write x_t / time columns of Abf (bf16)
//   Abf layout: [b][672]; cols 0..279 = x, 280 = tm, 281..664 = h (gate
//   kernel maintains these), 665 = tm, 666..671 = zero padding.
// ---------------------------------------------------------------------------
__global__ void pack_x(const float* __restrict__ x, const float* __restrict__ tim,
                       unsigned short* __restrict__ Abf, int t)
{
    int idx = blockIdx.x * blockDim.x + threadIdx.x;   // over 256*282
    if (idx >= Bsz * 282) return;
    int b = idx / 282;
    int j = idx % 282;
    int col = (j < 281) ? j : (KU - 1);
    float v = (j < Dd) ? x[((size_t)b * Tst + t) * Dd + j]
                       : tim[(size_t)b * Tst + t];
    Abf[(size_t)b * KP + col] = f32_to_bf16(v);
}

// ---------------------------------------------------------------------------
// Kernel 3 (per step): xo = Abf @ WB + bias  via bf16 WMMA, all operands
// loaded directly from global (L2-resident), fragments per documented
// 16-bit layouts (contiguous 16B runs per lane -> global_load_b128).
//   block = 128 threads (4 waves); wave w owns a 16x64 tile.
//   grid  = (Bsz/16, NP/256) = (16, 7)
// ---------------------------------------------------------------------------
__global__ void step_gemm(const unsigned short* __restrict__ Abf,
                          const unsigned short* __restrict__ WB,
                          const float* __restrict__ bias,
                          float* __restrict__ xo)
{
    const int lane = threadIdx.x & 31;
    const int wave = threadIdx.x >> 5;               // 0..3
    const int m0   = blockIdx.x * 16;                // batch-row tile
    const int n0   = blockIdx.y * 256 + wave * 64;   // this wave's N base

    const int mA   = lane & 15;
    const int half = lane >> 4;

    v8f acc0 = {}, acc1 = {}, acc2 = {}, acc3 = {};

    // A: lane (mA, half) reads rows of Abf; 16B runs at k0 = kk + 8*half
    const unsigned short* Arow = Abf + (size_t)(m0 + mA) * KP;

    for (int kk = 0; kk < KP; kk += 32) {
        FragU fa;
        {
            const uint4 alo = *(const uint4*)(Arow + kk + 8 * half);
            const uint4 ahi = *(const uint4*)(Arow + kk + 16 + 8 * half);
            fa.u[0] = alo.x; fa.u[1] = alo.y; fa.u[2] = alo.z; fa.u[3] = alo.w;
            fa.u[4] = ahi.x; fa.u[5] = ahi.y; fa.u[6] = ahi.z; fa.u[7] = ahi.w;
        }
        // B: lane holds column N (n0+j*16+mA); K in table order (8*half+s, 16+8*half+s-8)
        const unsigned short* Brow = WB + (size_t)(kk + 8 * half) * NP + n0 + mA;

        #pragma unroll
        for (int j = 0; j < 4; ++j) {
            FragU fb;
            #pragma unroll
            for (int p = 0; p < 8; ++p) {
                const int kr = (p < 4) ? 2 * p : 16 + 2 * (p - 4);
                const unsigned short* bp = Brow + j * 16 + (size_t)kr * NP;
                fb.u[p] = (unsigned int)bp[0] | ((unsigned int)bp[NP] << 16);
            }
            v8f& acc = (j == 0) ? acc0 : (j == 1) ? acc1 : (j == 2) ? acc2 : acc3;
            acc = __builtin_amdgcn_wmma_f32_16x16x32_bf16(
                      false, fa.v, false, fb.v, (short)0, acc, false, false);
        }
    }

    // store D + bias (C/D layout: vgpr r -> M = r + 8*half, lane%16 -> N)
    const int mrow = m0 + 8 * half;
    #pragma unroll
    for (int j = 0; j < 4; ++j) {
        const v8f& acc = (j == 0) ? acc0 : (j == 1) ? acc1 : (j == 2) ? acc2 : acc3;
        const int n = n0 + j * 16 + mA;
        const float bv = bias[n];
        #pragma unroll
        for (int r = 0; r < 8; ++r)
            xo[(size_t)(mrow + r) * NP + n] = acc[r] + bv;
    }
}

// ---------------------------------------------------------------------------
// Kernel 4 (per step): gating / state update. 1 block per batch, 384 threads.
// Also writes bf16 h into Abf cols 281..664 for the next step's GEMM.
// ---------------------------------------------------------------------------
__global__ void step_gate(const float* __restrict__ xo, float* __restrict__ c,
                          float* __restrict__ h, unsigned short* __restrict__ Abf,
                          float* __restrict__ h_ring, float* __restrict__ dist_ring,
                          int t)
{
    const int b   = blockIdx.x;
    const int tid = threadIdx.x;             // 0..383
    __shared__ float sfm[Ll];
    __shared__ float sim[Ll];
    __shared__ float sdist;

    const float* row = xo + (size_t)b * NP;

    if (tid == 0) {
        // fm = cumsum(softmax(row[0:L])), left-to-right
        float e[Ll];
        float mx = -1e30f;
        for (int l = 0; l < Ll; ++l) mx = fmaxf(mx, row[l]);
        float s = 0.0f;
        for (int l = 0; l < Ll; ++l) { e[l] = __expf(row[l] - mx); s += e[l]; }
        float inv = 1.0f / s, cs = 0.0f, fsum = 0.0f;
        for (int l = 0; l < Ll; ++l) { cs += e[l] * inv; sfm[l] = cs; fsum += cs; }
        sdist = 1.0f - fsum / (float)Ll;
        // im = reverse cumsum(softmax(row[L:2L])), right-to-left
        mx = -1e30f;
        for (int l = 0; l < Ll; ++l) mx = fmaxf(mx, row[Ll + l]);
        s = 0.0f;
        for (int l = 0; l < Ll; ++l) { e[l] = __expf(row[Ll + l] - mx); s += e[l]; }
        inv = 1.0f / s; cs = 0.0f;
        for (int l = Ll - 1; l >= 0; --l) { cs += e[l] * inv; sim[l] = cs; }
    }
    __syncthreads();

    const int l  = tid >> 5;
    const int ch = tid & 31;
    const float fm = sfm[l];
    const float im = sim[l];
    const float fg = sigmoidf_(row[2 * Ll + (0 * Ll + l) * CHUNK + ch]);
    const float ig = sigmoidf_(row[2 * Ll + (1 * Ll + l) * CHUNK + ch]);
    const float og = sigmoidf_(row[2 * Ll + (2 * Ll + l) * CHUNK + ch]);
    const float ci = tanhf   (row[2 * Ll + (3 * Ll + l) * CHUNK + ch]);

    const float cl = c[(size_t)b * Hh + tid];
    const float ov = fm * im;
    const float cn = ov * (fg * cl + ig * ci) + (fm - ov) * cl + (im - ov) * ci;
    const float hn = og * tanhf(cn);

    c[(size_t)b * Hh + tid] = cn;
    h[(size_t)b * Hh + tid] = hn;
    Abf[(size_t)b * KP + (Dd + 1) + tid] = f32_to_bf16(hn);   // next step's A

    if (t >= Tst - Kk) {
        const int kidx = t - (Tst - Kk);
        h_ring[((size_t)kidx * Bsz + b) * Hh + tid] = hn;
        if (tid == 0) dist_ring[kidx * Bsz + b] = sdist;
    }
}

// ---------------------------------------------------------------------------
// Kernel 5: pooling over last K steps, theme / conv / output. 1 block/batch.
// ---------------------------------------------------------------------------
__global__ void finalize(const float* __restrict__ h, const float* __restrict__ h_ring,
                         const float* __restrict__ dist_ring,
                         const float* __restrict__ Ws, const float* __restrict__ bs,
                         const float* __restrict__ Wrs, const float* __restrict__ brs,
                         const float* __restrict__ Wc, const float* __restrict__ bc,
                         const float* __restrict__ Wo, const float* __restrict__ bo,
                         float* __restrict__ out)
{
    const int b   = blockIdx.x;
    const int tid = threadIdx.x;          // 0..383
    __shared__ float ld[Kk];
    __shared__ float mh[Hh];
    __shared__ float s1[H6];
    __shared__ float red[Hh];

    if (tid == 0) {
        float tmp[Kk];
        float cum = 0.0f, mx = -1e30f;
        for (int k = 0; k < Kk; ++k) {
            cum += dist_ring[k * Bsz + b];
            tmp[k] = cum;
            mx = fmaxf(mx, cum);
        }
        float s = 0.0f;
        for (int k = 0; k < Kk; ++k) { tmp[k] = __expf(tmp[k] - mx); s += tmp[k]; }
        float inv = 1.0f / s;
        for (int k = 0; k < Kk; ++k) ld[k] = tmp[k] * inv;
    }
    __syncthreads();

    {   // mh[c] = mean_k local_h
        float acc = 0.0f;
        for (int k = 0; k < Kk; ++k)
            acc += h_ring[((size_t)k * Bsz + b) * Hh + tid] * ld[k];
        mh[tid] = acc / (float)Kk;
    }
    __syncthreads();

    if (tid < H6) {   // relu(mh @ Ws + bs)
        float acc = bs[tid];
        for (int cc = 0; cc < Hh; ++cc) acc += mh[cc] * Ws[cc * H6 + tid];
        s1[tid] = fmaxf(acc, 0.0f);
    }
    __syncthreads();

    float th;
    {   // theme = sigmoid(s1 @ Wrs + brs)
        float acc = brs[tid];
        for (int j = 0; j < H6; ++j) acc += s1[j] * Wrs[j * Hh + tid];
        th = sigmoidf_(acc);
    }

    // conv[o] = sum_{c,k} local_h[b][c][k] * Wc[o][c][k] + bc[o]
    float conv = bc[tid];
    for (int cc = 0; cc < Hh; ++cc) {
        float wsum = 0.0f;
        #pragma unroll
        for (int k = 0; k < Kk; ++k)
            wsum += h_ring[((size_t)k * Bsz + b) * Hh + cc] * ld[k]
                  * Wc[((size_t)tid * Hh + cc) * Kk + k];
        conv += wsum;
    }

    const float rnn = th * conv + h[(size_t)b * Hh + tid];
    red[tid] = rnn * Wo[tid];
    __syncthreads();

    if (tid == 0) {
        float s = bo[0];
        for (int i = 0; i < Hh; ++i) s += red[i];
        out[b] = sigmoidf_(s);
    }
}

// ---------------------------------------------------------------------------
// host launcher
// ---------------------------------------------------------------------------
extern "C" void kernel_launch(void* const* d_in, const int* in_sizes, int n_in,
                              void* d_out, int out_size, void* d_ws, size_t ws_size,
                              hipStream_t stream)
{
    (void)in_sizes; (void)n_in; (void)out_size; (void)ws_size;

    const float* x    = (const float*)d_in[0];
    const float* tim  = (const float*)d_in[1];
    const float* Wk   = (const float*)d_in[2];
    const float* bk   = (const float*)d_in[3];
    const float* Wr   = (const float*)d_in[4];
    const float* br   = (const float*)d_in[5];
    const float* Ws   = (const float*)d_in[6];
    const float* bs   = (const float*)d_in[7];
    const float* Wrs  = (const float*)d_in[8];
    const float* brs  = (const float*)d_in[9];
    const float* Wc   = (const float*)d_in[10];
    const float* bc   = (const float*)d_in[11];
    const float* Wo   = (const float*)d_in[12];
    const float* bo   = (const float*)d_in[13];
    float* out = (float*)d_out;

    // workspace carve-up (256B aligned)
    char* ws = (char*)d_ws;
    size_t off = 0;
    auto carve = [&](size_t bytes) -> void* {
        void* p = ws + off;
        off += (bytes + 255) & ~(size_t)255;
        return p;
    };
    unsigned short* WB   = (unsigned short*)carve((size_t)KP * NP * sizeof(unsigned short));
    float* bias          = (float*)carve((size_t)NP * sizeof(float));
    unsigned short* Abf  = (unsigned short*)carve((size_t)Bsz * KP * sizeof(unsigned short));
    float* xo            = (float*)carve((size_t)Bsz * NP * sizeof(float));
    float* hbuf          = (float*)carve((size_t)Bsz * Hh * sizeof(float));
    float* cbuf          = (float*)carve((size_t)Bsz * Hh * sizeof(float));
    float* h_ring        = (float*)carve((size_t)Kk * Bsz * Hh * sizeof(float));
    float* dist_ring     = (float*)carve((size_t)Kk * Bsz * sizeof(float));

    // 0) weights -> bf16 padded, combined bias
    {
        int total = KP * NP;
        prep_weights<<<(total + 255) / 256, 256, 0, stream>>>(Wk, bk, Wr, br, WB, bias);
    }
    // 1) zero h, c, Abf
    init_state<<<(Bsz * Hh + 255) / 256, 256, 0, stream>>>(hbuf, cbuf, (unsigned int*)Abf);

    // 2) sequential scan
    dim3 ggrid(Bsz / 16, NP / 256);   // 16 x 7
    const int packN = Bsz * 282;
    for (int t = 0; t < Tst; ++t) {
        pack_x   <<<(packN + 255) / 256, 256, 0, stream>>>(x, tim, Abf, t);
        step_gemm<<<ggrid, 128, 0, stream>>>(Abf, WB, bias, xo);
        step_gate<<<Bsz, Hh, 0, stream>>>(xo, cbuf, hbuf, Abf, h_ring, dist_ring, t);
    }

    // 3) final pooling / theme / conv / output
    finalize<<<Bsz, Hh, 0, stream>>>(hbuf, h_ring, dist_ring,
                                     Ws, bs, Wrs, brs, Wc, bc, Wo, bo, out);
}
